// GGNNSumNew_78151224918834
// MI455X (gfx1250) — matrix operand order repaired and checked by reading, not compiled, not run on hardware
//
#include <hip/hip_runtime.h>
#include <hip/hip_bf16.h>

typedef __attribute__((ext_vector_type(16))) _Float16 v16h;
typedef __attribute__((ext_vector_type(8)))  _Float16 v8h;
typedef __attribute__((ext_vector_type(4)))  _Float16 v4h;
typedef __attribute__((ext_vector_type(8)))  float    v8f;

#define NNODES 50000
#define NEDGES 800000
#define NGRAPH 100
#define DIM    128

// ---------------------------------------------------------------------------
// Elementwise helpers
// ---------------------------------------------------------------------------
__global__ void zero_f32_kernel(float* __restrict__ p, int n) {
    int i = blockIdx.x * blockDim.x + threadIdx.x;
    if (i < n) p[i] = 0.0f;
}

__global__ void init_h_kernel(const float* __restrict__ feat,
                              float* __restrict__ h, _Float16* __restrict__ h16, int N) {
    int idx = blockIdx.x * blockDim.x + threadIdx.x;
    if (idx >= N * DIM) return;
    int n = idx >> 7, d = idx & 127;
    float v = (d < 100) ? feat[(size_t)n * 100 + d] : 0.0f;
    h[idx] = v;
    h16[idx] = (_Float16)v;
}

// W_msg[t][d][f] (f32) -> Wmsg16[t][f][d] (f16)  (transpose so B-fragments are contiguous in k)
__global__ void conv_wmsg_kernel(const float* __restrict__ W, _Float16* __restrict__ out) {
    int idx = blockIdx.x * blockDim.x + threadIdx.x;
    if (idx >= 4 * DIM * DIM) return;
    int t = idx >> 14, rem = idx & 16383;
    int d = rem >> 7, f = rem & 127;
    out[(t << 14) | (f << 7) | d] = (_Float16)W[idx];
}

// plain f32 -> f16 convert (weights already in [out,in] layout)
__global__ void conv_f16_kernel(const float* __restrict__ W, _Float16* __restrict__ out, int n) {
    int i = blockIdx.x * blockDim.x + threadIdx.x;
    if (i < n) out[i] = (_Float16)W[i];
}

// ---------------------------------------------------------------------------
// WMMA GEMM:  C[m, o] = sum_k A[m,k] * Bt[o,k] + bias[o]
//   A  : [M x K]      f16, row-major, lda = K
//   Bt : [Ncols x K]  f16, row-major (i.e. B transposed)
//   C  : [M x Ncols]  f32 or f16 (OUT_F16)
// Each wave computes 5 consecutive 16x16 m-tiles for one n-tile (M = 80 * mgroups).
// gridDim.z selects a slice via strideB / strideBias / strideC.
// ---------------------------------------------------------------------------
template <bool OUT_F16>
__global__ void gemm_wmma_kernel(const _Float16* __restrict__ A,
                                 const _Float16* __restrict__ Bt,
                                 const float* __restrict__ bias,
                                 void* __restrict__ Cout,
                                 int M, int K, int Ncols,
                                 long strideB, long strideBias, long strideC) {
    const int z = blockIdx.z;
    Bt   += (size_t)z * strideB;
    bias += (size_t)z * strideBias;

    const int lane  = threadIdx.x & 31;
    const int wave  = threadIdx.x >> 5;
    const int ntN   = Ncols >> 4;
    const int tid   = blockIdx.x * (blockDim.x >> 5) + wave;
    const int mgrp  = tid / ntN;
    const int ntile = tid % ntN;
    const int m0    = mgrp * 80;
    if (m0 >= M) return;                 // wave-uniform; EXEC stays all-ones

    const int hs   = lane >> 4;          // half-wave select
    const int lrow = lane & 15;

    v8f zero = {};
    v8f acc[5];
    #pragma unroll
    for (int i = 0; i < 5; ++i) acc[i] = zero;

    // B fragment: col = ntile*16 + lrow, 16 consecutive k starting at hs*16
    const _Float16* bptr = Bt + (size_t)(ntile * 16 + lrow) * K + hs * 16;
    // A fragment: row = m0 + i*16 + lrow, 16 consecutive k starting at hs*16
    const _Float16* aptr = A + (size_t)(m0 + lrow) * K + hs * 16;

    for (int kk = 0; kk < K; kk += 32) {
        union { v16h v; v8h h[2]; } bu;
        bu.v = *(const v16h*)(bptr + kk);
        #pragma unroll
        for (int i = 0; i < 5; ++i) {
            union { v16h v; v8h h[2]; } au;
            const _Float16* ap = aptr + (size_t)i * 16 * K + kk;
            au.h[0] = *(const v8h*)(ap);
            au.h[1] = *(const v8h*)(ap + 8);
            acc[i] = __builtin_amdgcn_wmma_f32_16x16x32_f16(
                false, au.v, false, bu.v, (short)0, acc[i], false, false);
        }
    }

    const int  ncol = ntile * 16 + lrow;
    const float bv  = bias ? bias[ncol] : 0.0f;
    #pragma unroll
    for (int i = 0; i < 5; ++i) {
        const int mbase = m0 + i * 16 + hs * 8;
        #pragma unroll
        for (int r = 0; r < 8; ++r) {
            float v = acc[i][r] + bv;
            size_t off = (size_t)z * strideC + (size_t)(mbase + r) * Ncols + ncol;
            if (OUT_F16) ((_Float16*)Cout)[off] = (_Float16)v;
            else         ((float*)Cout)[off]   = v;
        }
    }
}

// ---------------------------------------------------------------------------
// Edge gather + scatter-add: one wave per edge, 4 features per lane.
// T16 is L2-resident (51 MB), agg is L2-resident (25 MB).
// ---------------------------------------------------------------------------
__global__ void scatter_edges_kernel(const int* __restrict__ esrc,
                                     const int* __restrict__ edst,
                                     const int* __restrict__ etyp,
                                     const _Float16* __restrict__ T16,
                                     float* __restrict__ agg, int E, int N) {
    int gtid = blockIdx.x * blockDim.x + threadIdx.x;
    int e    = gtid >> 5;
    int lane = threadIdx.x & 31;
    if (e >= E) return;
    int s = esrc[e], d = edst[e], t = etyp[e];
    const _Float16* m = T16 + (((size_t)t * N + s) << 7) + lane * 4;
    float*          o = agg + ((size_t)d << 7) + lane * 4;
    v4h mv = *(const v4h*)m;
    atomicAdd(o + 0, (float)mv[0]);
    atomicAdd(o + 1, (float)mv[1]);
    atomicAdd(o + 2, (float)mv[2]);
    atomicAdd(o + 3, (float)mv[3]);
}

__global__ void f32_to_f16_kernel(const float* __restrict__ in, _Float16* __restrict__ out, int n) {
    int i = blockIdx.x * blockDim.x + threadIdx.x;
    if (i < n) out[i] = (_Float16)in[i];
}

// ---------------------------------------------------------------------------
// GRU combine (torch GRUCell gate order r,z,n): reads gi/gh [N,384], updates h.
// ---------------------------------------------------------------------------
__global__ void gru_update_kernel(const float* __restrict__ gi,
                                  const float* __restrict__ gh,
                                  float* __restrict__ h, _Float16* __restrict__ h16, int N) {
    int idx = blockIdx.x * blockDim.x + threadIdx.x;
    if (idx >= N * DIM) return;
    int n = idx >> 7, d = idx & 127;
    const float* gin = gi + (size_t)n * 384;
    const float* ghn = gh + (size_t)n * 384;
    float ir = gin[d],       hr = ghn[d];
    float iz = gin[128 + d], hz = ghn[128 + d];
    float in_ = gin[256 + d], hn = ghn[256 + d];
    float r  = 1.0f / (1.0f + __expf(-(ir + hr)));
    float z  = 1.0f / (1.0f + __expf(-(iz + hz)));
    float nn = tanhf(in_ + r * hn);
    float out = (1.0f - z) * nn + z * h[idx];
    h[idx]   = out;
    h16[idx] = (_Float16)out;
}

// ---------------------------------------------------------------------------
// Per-graph pooling: one wave per node, 4 features per lane.
// ---------------------------------------------------------------------------
__global__ void pool_accum_kernel(const float* __restrict__ h,
                                  const int* __restrict__ gid,
                                  float* __restrict__ pooled,
                                  float* __restrict__ counts, int N) {
    int gtid = blockIdx.x * blockDim.x + threadIdx.x;
    int n    = gtid >> 5;
    int lane = threadIdx.x & 31;
    if (n >= N) return;
    int g = gid[n];
    const float* hp = h + ((size_t)n << 7) + lane * 4;
    float*       pp = pooled + ((size_t)g << 7) + lane * 4;
    atomicAdd(pp + 0, hp[0]);
    atomicAdd(pp + 1, hp[1]);
    atomicAdd(pp + 2, hp[2]);
    atomicAdd(pp + 3, hp[3]);
    if (lane == 0) atomicAdd(counts + g, 1.0f);
}

// ---------------------------------------------------------------------------
// Classifier: one block per graph; mean -> Linear(128->256) -> relu -> dot(256) -> sigmoid
// ---------------------------------------------------------------------------
__global__ void classifier_kernel(const float* __restrict__ pooled,
                                  const float* __restrict__ counts,
                                  const float* __restrict__ W1, const float* __restrict__ b1,
                                  const float* __restrict__ W2, const float* __restrict__ b2,
                                  float* __restrict__ out) {
    int g = blockIdx.x;
    int j = threadIdx.x;               // 256 threads = hidden dim
    __shared__ float pl[DIM];
    __shared__ float red[256];
    float cnt = fmaxf(counts[g], 1.0f);
    if (j < DIM) pl[j] = pooled[(size_t)g * DIM + j] / cnt;
    __syncthreads();
    float acc = b1[j];
    #pragma unroll 4
    for (int k = 0; k < DIM; ++k) acc += pl[k] * W1[(size_t)j * DIM + k];
    float hid = fmaxf(acc, 0.0f);
    red[j] = hid * W2[j];
    __syncthreads();
    for (int s = 128; s > 0; s >>= 1) {
        if (j < s) red[j] += red[j + s];
        __syncthreads();
    }
    if (j == 0) {
        float lg = red[0] + b2[0];
        out[g] = 1.0f / (1.0f + __expf(-lg));
    }
}

// ---------------------------------------------------------------------------
// Host orchestration
// ---------------------------------------------------------------------------
extern "C" void kernel_launch(void* const* d_in, const int* in_sizes, int n_in,
                              void* d_out, int out_size, void* d_ws, size_t ws_size,
                              hipStream_t stream) {
    const float* features   = (const float*)d_in[0];
    const int*   edge_src   = (const int*)d_in[1];
    const int*   edge_dst   = (const int*)d_in[2];
    const int*   edge_types = (const int*)d_in[3];
    const int*   graph_ids  = (const int*)d_in[4];
    const float* W_msg      = (const float*)d_in[5];
    const float* b_msg      = (const float*)d_in[6];
    const float* W_ih       = (const float*)d_in[7];
    const float* W_hh       = (const float*)d_in[8];
    const float* b_ih       = (const float*)d_in[9];
    const float* b_hh       = (const float*)d_in[10];
    const float* W1         = (const float*)d_in[11];
    const float* b1         = (const float*)d_in[12];
    const float* W2         = (const float*)d_in[13];
    const float* b2         = (const float*)d_in[14];
    float* out = (float*)d_out;

    const int N = NNODES, E = NEDGES, G = NGRAPH;
    char* ws = (char*)d_ws;
    size_t off = 0;
    auto alloc = [&](size_t bytes) -> char* {
        char* p = ws + off;
        off += (bytes + 255) & ~(size_t)255;
        return p;
    };

    float*    h      = (float*)   alloc((size_t)N * DIM * 4);          // 25.6 MB
    _Float16* h16    = (_Float16*)alloc((size_t)N * DIM * 2);          // 12.8 MB
    float*    a      = (float*)   alloc((size_t)N * DIM * 4);          // 25.6 MB
    _Float16* a16    = (_Float16*)alloc((size_t)N * DIM * 2);          // 12.8 MB
    // shared region: T16 (4*N*128 f16 = 51.2 MB) overlaps gi+gh (2*N*384 f32 = 153.6 MB);
    // T16 is fully consumed by scatter before the gate GEMMs overwrite it.
    char*     region = alloc((size_t)2 * N * 384 * 4);                 // 153.6 MB
    _Float16* T16    = (_Float16*)region;
    float*    gi     = (float*)region;
    float*    gh     = (float*)(region + (size_t)N * 384 * 4);
    _Float16* Wmsg16 = (_Float16*)alloc((size_t)4 * DIM * DIM * 2);
    _Float16* Wih16  = (_Float16*)alloc((size_t)384 * DIM * 2);
    _Float16* Whh16  = (_Float16*)alloc((size_t)384 * DIM * 2);
    float*    pooled = (float*)   alloc((size_t)G * DIM * 4 + G * 4);  // pooled + counts contiguous
    float*    counts = pooled + (size_t)G * DIM;

    // --- one-time conversions / init ---
    conv_wmsg_kernel<<<(4 * DIM * DIM + 255) / 256, 256, 0, stream>>>(W_msg, Wmsg16);
    conv_f16_kernel<<<(384 * DIM + 255) / 256, 256, 0, stream>>>(W_ih, Wih16, 384 * DIM);
    conv_f16_kernel<<<(384 * DIM + 255) / 256, 256, 0, stream>>>(W_hh, Whh16, 384 * DIM);
    init_h_kernel<<<(N * DIM + 255) / 256, 256, 0, stream>>>(features, h, h16, N);

    // GEMM launch geometry: M = 50000 = 625 * 80 (5 m-tiles per wave), 8 waves/block
    const int mgroups = N / 80;                  // 625
    const int blkMsg  = mgroups * (DIM / 16) / 8;   // 625  (Ncols=128 -> 8 n-tiles)
    const int blkGate = mgroups * (384 / 16) / 8;   // 1875 (Ncols=384 -> 24 n-tiles)

    for (int step = 0; step < 8; ++step) {
        // (1) per-etype message transform: T16[t] = h @ W_msg[t] + b_msg[t]   (f16 out)
        gemm_wmma_kernel<true><<<dim3(blkMsg, 1, 4), 256, 0, stream>>>(
            h16, Wmsg16, b_msg, (void*)T16, N, DIM, DIM,
            (long)DIM * DIM, (long)DIM, (long)N * DIM);

        // (2) zero accumulator, gather + scatter-add over edges
        zero_f32_kernel<<<(N * DIM + 255) / 256, 256, 0, stream>>>(a, N * DIM);
        scatter_edges_kernel<<<(E * 32 + 255) / 256, 256, 0, stream>>>(
            edge_src, edge_dst, edge_types, T16, a, E, N);

        // (3) GRU gate GEMMs (f32 out): gi = a @ W_ih^T + b_ih ; gh = h @ W_hh^T + b_hh
        f32_to_f16_kernel<<<(N * DIM + 255) / 256, 256, 0, stream>>>(a, a16, N * DIM);
        gemm_wmma_kernel<false><<<dim3(blkGate, 1, 1), 256, 0, stream>>>(
            a16, Wih16, b_ih, (void*)gi, N, DIM, 384, 0, 0, 0);
        gemm_wmma_kernel<false><<<dim3(blkGate, 1, 1), 256, 0, stream>>>(
            h16, Whh16, b_hh, (void*)gh, N, DIM, 384, 0, 0, 0);

        // (4) GRU elementwise combine -> new h (+ f16 shadow)
        gru_update_kernel<<<(N * DIM + 255) / 256, 256, 0, stream>>>(gi, gh, h, h16, N);
    }

    // --- pooling + classifier ---
    zero_f32_kernel<<<(G * DIM + G + 255) / 256, 256, 0, stream>>>(pooled, G * DIM + G);
    pool_accum_kernel<<<(N * 32 + 255) / 256, 256, 0, stream>>>(h, graph_ids, pooled, counts, N);
    classifier_kernel<<<G, 256, 0, stream>>>(pooled, counts, W1, b1, W2, b2, out);
}
